// CNL_59356448031474
// MI455X (gfx1250) — matrix operand order repaired and checked by reading, not compiled, not run on hardware
//
#include <hip/hip_runtime.h>
#include <hip/hip_bf16.h>

typedef __attribute__((ext_vector_type(16))) _Float16 v16h;
typedef __attribute__((ext_vector_type(8)))  _Float16 v8h;
typedef __attribute__((ext_vector_type(8)))  float    v8f;
typedef int i4v __attribute__((vector_size(16)));

constexpr int CB = 32;        // batch
constexpr int CC = 512;       // dim (channels)
constexpr int CN = 48 * 24;   // spatial N = 1152
constexpr int BM = 128, BN = 128, BK = 32;
constexpr int LROW = BK + 8;  // padded LDS row (halves) -> 80B

union FragH { v16h v; v8h h[2]; };

// ---- CDNA5 async global->LDS copy (16B per lane), ASYNCcnt-tracked ----
__device__ __forceinline__ void async_cp16(const _Float16* g, _Float16* l) {
#if __has_builtin(__builtin_amdgcn_global_load_async_to_lds_b128)
  __builtin_amdgcn_global_load_async_to_lds_b128(
      (__attribute__((address_space(1))) i4v*)(i4v*)g,
      (__attribute__((address_space(3))) i4v*)(i4v*)l, 0, 0);
#else
  unsigned lo = (unsigned)(unsigned long)(__attribute__((address_space(3))) void*)l;
  asm volatile("global_load_async_to_lds_b128 %0, %1, off"
               :: "v"(lo), "v"(g) : "memory");
#endif
}

__device__ __forceinline__ void wait_async0() {
#if __has_builtin(__builtin_amdgcn_s_wait_asynccnt)
  __builtin_amdgcn_s_wait_asynccnt(0);
#else
  asm volatile("s_wait_asynccnt 0" ::: "memory");
#endif
}

// ---------------- converts ----------------

__global__ __launch_bounds__(256) void cvt_w_f16(const float* __restrict__ in,
                                                 _Float16* __restrict__ out, int n) {
  int i = blockIdx.x * 256 + threadIdx.x;
  if (i < n) out[i] = (_Float16)in[i];
}

// in: [b][C][N] f32  ->  out: [b][N][C] f16  (K-contiguous rows for B operand)
__global__ __launch_bounds__(256) void cvt_xT_f16(const float* __restrict__ in,
                                                  _Float16* __restrict__ out) {
  int i = blockIdx.x * 256 + threadIdx.x;  // over N*C, c fastest
  int b = blockIdx.y;
  if (i < CC * CN) {
    int c = i % CC;
    int n = i / CC;
    out[(long)b * CC * CN + i] = (_Float16)in[(long)b * CC * CN + (long)c * CN + n];
  }
}

// ---------------- generic WMMA GEMM ----------------
// C[row,col] = sum_k A[row,k] * Bt[col,k]   (Bt stored K-contiguous)
// MODE 0: +bias[row], store f16 [row*ldc+col]            (theta, phi)
// MODE 1: +bias[row], store f16 transposed [col*ldc+row] (g)
// MODE 2: *1/512, store f16 [row*ldc+col]                (attention)
// MODE 3: store f16 transposed [col*ldc+row]             (y)
// MODE 4: +bias, BN(gamma,beta,mean,var) + residual, store f32 (final)

template <int MODE>
__global__ __launch_bounds__(256) void gemm_wmma(
    const _Float16* __restrict__ A, int lda, long sA,
    const _Float16* __restrict__ Bt, int ldb, long sB,
    void* __restrict__ Cout, int ldc, long sC,
    int K,
    const float* __restrict__ bias,
    const float* __restrict__ gma, const float* __restrict__ bta,
    const float* __restrict__ rmean, const float* __restrict__ rvar,
    const float* __restrict__ resid, long sR) {
  __shared__ __align__(16) _Float16 As[2][BM][LROW];
  __shared__ __align__(16) _Float16 Bs[2][BN][LROW];

  const int b    = blockIdx.z;
  const int tid  = threadIdx.x;
  const int lane = tid & 31;
  const int wid  = tid >> 5;   // 8 waves
  const int wM   = wid & 3;    // 4 wave-rows  (32 rows each)
  const int wN   = wid >> 2;   // 2 wave-cols  (64 cols each)
  const int lm   = lane & 15;
  const int hh   = lane >> 4;  // lane half (ISA K-split / M-split)

  const _Float16* Ab = A  + (long)b * sA;
  const _Float16* Bb = Bt + (long)b * sB;
  const int rowBase = blockIdx.y * BM;
  const int colBase = blockIdx.x * BN;

  // per-thread staging slots: 256 threads x 2 chunks cover 128 rows x 32 halves
  const int r0 = tid >> 2;            // 0..63
  const int s0 = (tid & 3) * 8;       // 16B chunk
  const _Float16* gA0 = Ab + (long)(rowBase + r0) * lda + s0;
  const _Float16* gA1 = Ab + (long)(rowBase + r0 + 64) * lda + s0;
  const _Float16* gB0 = Bb + (long)(colBase + r0) * ldb + s0;
  const _Float16* gB1 = Bb + (long)(colBase + r0 + 64) * ldb + s0;

  auto stage = [&](int buf, int k0) {
    async_cp16(gA0 + k0, &As[buf][r0][s0]);
    async_cp16(gA1 + k0, &As[buf][r0 + 64][s0]);
    async_cp16(gB0 + k0, &Bs[buf][r0][s0]);
    async_cp16(gB1 + k0, &Bs[buf][r0 + 64][s0]);
  };

  v8f acc[2][4];
#pragma unroll
  for (int i = 0; i < 2; ++i)
#pragma unroll
    for (int j = 0; j < 4; ++j)
#pragma unroll
      for (int e = 0; e < 8; ++e) acc[i][j][e] = 0.f;

  const int nk = K / BK;
  stage(0, 0);

  for (int i = 0; i < nk; ++i) {
    const int buf = i & 1;
    wait_async0();        // this wave's copies into As/Bs[buf] have landed
    __syncthreads();      // all waves landed; prev compute done -> buf^1 reusable
    if (i + 1 < nk) stage(buf ^ 1, (i + 1) * BK);

    // A fragment (16x32 f16): lane half hh selects K {hh*8..+7, 16+hh*8..+7}
    FragH af[2], bf[4];
#pragma unroll
    for (int tm = 0; tm < 2; ++tm) {
      const _Float16* ap = &As[buf][wM * 32 + tm * 16 + lm][0];
      af[tm].h[0] = *(const v8h*)(ap + hh * 8);
      af[tm].h[1] = *(const v8h*)(ap + 16 + hh * 8);
    }
    // B fragment (32x16 f16): lane half hh selects K {hh*8..+7, 16+hh*8..+7}
#pragma unroll
    for (int tn = 0; tn < 4; ++tn) {
      const _Float16* bp = &Bs[buf][wN * 64 + tn * 16 + lm][0];
      bf[tn].h[0] = *(const v8h*)(bp + hh * 8);
      bf[tn].h[1] = *(const v8h*)(bp + 16 + hh * 8);
    }
#pragma unroll
    for (int tm = 0; tm < 2; ++tm)
#pragma unroll
      for (int tn = 0; tn < 4; ++tn)
        acc[tm][tn] = __builtin_amdgcn_wmma_f32_16x16x32_f16(
            false, af[tm].v, false, bf[tn].v, (short)0, acc[tm][tn], false, false);
  }

  // epilogue: C layout -> VGPR r, lane: M = r + hh*8, N = lm
#pragma unroll
  for (int tm = 0; tm < 2; ++tm) {
#pragma unroll
    for (int tn = 0; tn < 4; ++tn) {
#pragma unroll
      for (int r = 0; r < 8; ++r) {
        int row = rowBase + wM * 32 + tm * 16 + hh * 8 + r;
        int col = colBase + wN * 64 + tn * 16 + lm;
        float v = acc[tm][tn][r];
        if (MODE == 0 || MODE == 1 || MODE == 4) v += bias[row];
        if (MODE == 2) v *= (1.0f / 512.0f);
        if (MODE == 4) {
          float inv = gma[row] * rsqrtf(rvar[row] + 1e-5f);
          v = (v - rmean[row]) * inv + bta[row] +
              resid[(long)b * sR + (long)row * ldc + col];
          ((float*)Cout)[(long)b * sC + (long)row * ldc + col] = v;
        } else if (MODE == 1 || MODE == 3) {
          ((_Float16*)Cout)[(long)b * sC + (long)col * ldc + row] = (_Float16)v;
        } else {
          ((_Float16*)Cout)[(long)b * sC + (long)row * ldc + col] = (_Float16)v;
        }
      }
    }
  }
}

// ---------------- launcher ----------------

extern "C" void kernel_launch(void* const* d_in, const int* in_sizes, int n_in,
                              void* d_out, int out_size, void* d_ws, size_t ws_size,
                              hipStream_t stream) {
  (void)in_sizes; (void)n_in; (void)out_size; (void)ws_size;
  const float* x     = (const float*)d_in[0];
  const float* xh    = (const float*)d_in[1];
  const float* Wg    = (const float*)d_in[2];
  const float* bg    = (const float*)d_in[3];
  const float* Wt    = (const float*)d_in[4];
  const float* bt    = (const float*)d_in[5];
  const float* Wp    = (const float*)d_in[6];
  const float* bp    = (const float*)d_in[7];
  const float* Ww    = (const float*)d_in[8];
  const float* bw    = (const float*)d_in[9];
  const float* gma   = (const float*)d_in[10];
  const float* bta   = (const float*)d_in[11];
  const float* rmean = (const float*)d_in[12];
  const float* rvar  = (const float*)d_in[13];
  float* out = (float*)d_out;

  const long PB = (long)CC * CN;  // 589824 per-batch elems

  char* ws = (char*)d_ws;
  size_t off = 0;
  auto take = [&](size_t bytes) {
    void* p = ws + off;
    off += (bytes + 255) & ~(size_t)255;
    return p;
  };

  _Float16* xT   = (_Float16*)take((size_t)CB * PB * 2);       // [b][n][c]
  _Float16* xhT  = (_Float16*)take((size_t)CB * PB * 2);       // [b][n][c]
  _Float16* Wg16 = (_Float16*)take((size_t)CC * CC * 2);
  _Float16* Wt16 = (_Float16*)take((size_t)CC * CC * 2);
  _Float16* Wp16 = (_Float16*)take((size_t)CC * CC * 2);
  _Float16* Ww16 = (_Float16*)take((size_t)CC * CC * 2);
  _Float16* gT   = (_Float16*)take((size_t)CB * PB * 2);       // [b][n][j]
  _Float16* th   = (_Float16*)take((size_t)CB * PB * 2);       // [b][i][n]
  _Float16* ph   = (_Float16*)take((size_t)CB * PB * 2);       // [b][j][n]
  _Float16* at   = (_Float16*)take((size_t)CB * CC * CC * 2);  // [b][i][j]
  _Float16* yT   = (_Float16*)take((size_t)CB * PB * 2);       // [b][n][i]

  cvt_w_f16<<<(CC * CC + 255) / 256, 256, 0, stream>>>(Wg, Wg16, CC * CC);
  cvt_w_f16<<<(CC * CC + 255) / 256, 256, 0, stream>>>(Wt, Wt16, CC * CC);
  cvt_w_f16<<<(CC * CC + 255) / 256, 256, 0, stream>>>(Wp, Wp16, CC * CC);
  cvt_w_f16<<<(CC * CC + 255) / 256, 256, 0, stream>>>(Ww, Ww16, CC * CC);
  dim3 cg((CC * CN + 255) / 256, CB);
  cvt_xT_f16<<<cg, 256, 0, stream>>>(x, xT);
  cvt_xT_f16<<<cg, 256, 0, stream>>>(xh, xhT);

  dim3 blk(256);
  dim3 gP(CN / BN, CC / BM, CB);  // (9,4,32)  512x1152 outputs
  dim3 gE(CC / BN, CC / BM, CB);  // (4,4,32)  512x512 outputs
  long sAt = (long)CC * CC;

  // theta = Wt . x + bt            [i][n]
  gemm_wmma<0><<<gP, blk, 0, stream>>>(Wt16, CC, 0, xT, CC, PB, th, CN, PB, CC,
                                       bt, nullptr, nullptr, nullptr, nullptr, nullptr, 0);
  // phi = Wp . xh + bp             [j][n]
  gemm_wmma<0><<<gP, blk, 0, stream>>>(Wp16, CC, 0, xhT, CC, PB, ph, CN, PB, CC,
                                       bp, nullptr, nullptr, nullptr, nullptr, nullptr, 0);
  // g = Wg . xh + bg, stored gT    [n][j]
  gemm_wmma<1><<<gP, blk, 0, stream>>>(Wg16, CC, 0, xhT, CC, PB, gT, CC, PB, CC,
                                       bg, nullptr, nullptr, nullptr, nullptr, nullptr, 0);
  // attn = (theta . phi^T)/512     [i][j], K = N = 1152
  gemm_wmma<2><<<gE, blk, 0, stream>>>(th, CN, PB, ph, CN, PB, at, CC, sAt, CN,
                                       nullptr, nullptr, nullptr, nullptr, nullptr, nullptr, 0);
  // y = attn . g, stored yT        [n][i], K = 512
  gemm_wmma<3><<<gP, blk, 0, stream>>>(at, CC, sAt, gT, CC, PB, yT, CC, PB, CC,
                                       nullptr, nullptr, nullptr, nullptr, nullptr, nullptr, 0);
  // out = BN(Ww . y + bw) + x      f32
  gemm_wmma<4><<<gP, blk, 0, stream>>>(Ww16, CC, 0, yT, CC, PB, out, CN, PB, CC,
                                       bw, gma, bta, rmean, rvar, x, PB);
}
